// Quantizer_4131758539405
// MI455X (gfx1250) — matrix operand order repaired and checked by reading, not compiled, not run on hardware
//
#include <hip/hip_runtime.h>
#include <hip/hip_bf16.h>

typedef __attribute__((ext_vector_type(16))) __bf16 v16bf;
typedef __attribute__((ext_vector_type(8)))  float  v8f;
typedef __attribute__((ext_vector_type(4)))  unsigned int v4u;
typedef __attribute__((ext_vector_type(8)))  int v8i_t;
typedef __attribute__((ext_vector_type(4)))  int v4i_t;

union V16 { uint4 u[2]; v16bf v; };

// --- bf16 split helpers (RNE) ---
__device__ __forceinline__ unsigned short f2bf(float x) {
    unsigned u = __builtin_bit_cast(unsigned, x);
    unsigned r = u + 0x7FFFu + ((u >> 16) & 1u);
    return (unsigned short)(r >> 16);
}
__device__ __forceinline__ float bf2f(unsigned short h) {
    unsigned u = ((unsigned)h) << 16;
    return __builtin_bit_cast(float, u);
}

#define NROWS   32768          // 8 * 16^3
#define NCODES  4096
#define KDIM    256
#define QELEMS  8388608        // 8*256*4096
#define ENC_OFF (2*QELEMS)
#define DIFF_OFF (2*QELEMS + NROWS)

// LDS staging geometry: 16 codebook rows x 256 K bf16, padded row stride 528 B
#define ROW_STRIDE 528         // 512 B data + 16 B TDM pad (bank decorrelation)
#define TILE_B     (16 * ROW_STRIDE)   // 8448 B per hi or lo tile
#define STAGE_B    (2 * TILE_B)        // 16896 B per stage (hi + lo)

// ---------------------------------------------------------------------------
// TDM: DMA one 16x256 u16 tile (one codebook group) from global into LDS.
// 2D descriptor; pad_interval=6 (128 DW = one 512B row), pad_amount=3 (4 DW)
// => rows land at 528 B stride in LDS.
// ---------------------------------------------------------------------------
__device__ __forceinline__ void tdm_load_tile(const unsigned short* gsrc,
                                              unsigned lds_off) {
    unsigned long long ga = (unsigned long long)(uintptr_t)gsrc;
    v4u g0;
    g0[0] = 1u;                                    // count=1, no gather
    g0[1] = lds_off;                               // LDS byte address
    g0[2] = (unsigned)(ga & 0xFFFFFFFFu);          // global addr [31:0]
    g0[3] = (unsigned)((ga >> 32) & 0x01FFFFFFu)   // global addr [56:32]
            | 0x80000000u;                         // type=2 ("image")
    v8i_t g1;
    g1[0] = 0x07910000;        // data_size=1(2B) | pad_en | interval=6 | amount=3
    g1[1] = (int)(256u << 16); // tensor_dim0 = 256 (low16 in [31:16])
    g1[2] = (int)(4096u << 16);// tensor_dim1 = 4096 (low16 in [31:16]); dim0 hi16=0
    g1[3] = (int)(256u << 16); // tile_dim0 = 256; tensor_dim1 hi16 = 0
    g1[4] = 16;                // tile_dim1 = 16; tile_dim2 = 0
    g1[5] = 256;               // tensor_dim0_stride = 256 elements
    g1[6] = 0;
    g1[7] = 0;
    v4i_t z4 = {0, 0, 0, 0};
    v8i_t z8 = {0, 0, 0, 0, 0, 0, 0, 0};
    __builtin_amdgcn_tensor_load_to_lds(g0, g1, z4, z4, z8, 0);
}

// ---------------------------------------------------------------------------
// 1) Split weight -> bf16 hi/lo, compute |w|^2 per code, zero diff accumulator
// ---------------------------------------------------------------------------
__global__ void prep_weight(const float* __restrict__ w,
                            unsigned short* __restrict__ w_hi,
                            unsigned short* __restrict__ w_lo,
                            float* __restrict__ wnorm,
                            float* __restrict__ sumsq) {
    __shared__ float red[256];
    const int v = blockIdx.x, k = threadIdx.x;
    const size_t o = (size_t)v * KDIM + k;
    float x = w[o];
    unsigned short hb = f2bf(x);
    unsigned short lb = f2bf(x - bf2f(hb));
    w_hi[o] = hb;
    w_lo[o] = lb;
    red[k] = x * x;
    __syncthreads();
    for (int off = 128; off > 0; off >>= 1) {
        if (k < off) red[k] += red[k + off];
        __syncthreads();
    }
    if (k == 0) wnorm[v] = red[0];
    if (v == 0 && k == 0) *sumsq = 0.0f;
}

// ---------------------------------------------------------------------------
// 2) Transpose grid_feat [b,c,s] -> row-major [n=b*4096+s, k=c], split bf16
// ---------------------------------------------------------------------------
__global__ void prep_feat(const float* __restrict__ grid,
                          unsigned short* __restrict__ f_hi,
                          unsigned short* __restrict__ f_lo) {
    __shared__ float tile[32][33];
    const int tx = threadIdx.x & 31;
    const int ty = threadIdx.x >> 5;            // 0..7
    const int sT = (blockIdx.x & 127) * 32;     // 128 s-tiles
    const int cT = ((blockIdx.x >> 7) & 7) * 32;// 8 c-tiles
    const int b  = blockIdx.x >> 10;            // 8 batches
#pragma unroll
    for (int i = 0; i < 4; ++i) {
        int c = cT + ty + i * 8;
        tile[ty + i * 8][tx] = grid[((size_t)(b * 256 + c) << 12) + sT + tx];
    }
    __syncthreads();
#pragma unroll
    for (int i = 0; i < 4; ++i) {
        int s = sT + ty + i * 8;
        int c = cT + tx;
        float x = tile[tx][ty + i * 8];
        unsigned short hb = f2bf(x);
        unsigned short lb = f2bf(x - bf2f(hb));
        size_t o = ((size_t)((b << 12) + s) << 8) + c;
        f_hi[o] = hb;
        f_lo[o] = lb;
    }
}

// ---------------------------------------------------------------------------
// 3) Fused GEMM + argmin. One wave owns 16 rows; A fragments (hi+lo) stay in
//    registers across the whole 4096-code loop. B tiles are staged into LDS
//    by the Tensor Data Mover, double-buffered: wave 0 issues the DMA for
//    tile t+1 while all 8 waves run 24 WMMAs on tile t from LDS.
// ---------------------------------------------------------------------------
__global__ __launch_bounds__(256, 1)
void argmin_gemm(const unsigned short* __restrict__ f_hi,
                 const unsigned short* __restrict__ f_lo,
                 const unsigned short* __restrict__ w_hi,
                 const unsigned short* __restrict__ w_lo,
                 const float* __restrict__ wnorm,
                 int* __restrict__ idx_out,
                 float* __restrict__ enc_out) {
    __shared__ __align__(16) char ldsb[2][STAGE_B];   // double-buffered B stage

    const int lane = threadIdx.x & 31;
    const int wave = blockIdx.x * 8 + (threadIdx.x >> 5);
    const int r0   = wave * 16;
    const int h    = lane >> 4;
    const int l15  = lane & 15;
    const bool issuer = (threadIdx.x >> 5) == 0;      // wave 0 drives the TDM

    // Register-resident A fragments (16-bit A 16x32 lane layout).
    v16bf a_hi[8], a_lo[8];
    const size_t rowBase = (size_t)(r0 + l15) * KDIM + h * 8;
#pragma unroll
    for (int ks = 0; ks < 8; ++ks) {
        const size_t base = rowBase + ks * 32;
        V16 t;
        t.u[0] = *(const uint4*)(f_hi + base);
        t.u[1] = *(const uint4*)(f_hi + base + 16);
        a_hi[ks] = t.v;
        t.u[0] = *(const uint4*)(f_lo + base);
        t.u[1] = *(const uint4*)(f_lo + base + 16);
        a_lo[ks] = t.v;
    }

    const unsigned lds_base = (unsigned)(uintptr_t)&ldsb[0][0];

    if (issuer) {   // prime stage 0 with tile 0
        tdm_load_tile(w_hi, lds_base);
        tdm_load_tile(w_lo, lds_base + TILE_B);
    }

    float bestVal[8];
    int   bestIdx[8];
#pragma unroll
    for (int i = 0; i < 8; ++i) { bestVal[i] = 3.4e38f; bestIdx[i] = 0; }

    const unsigned cOff = (unsigned)(l15 * ROW_STRIDE + h * 16); // bytes

    for (int t = 0; t < 256; ++t) {
        __syncthreads();   // everyone done reading buffer (t+1)&1 from iter t-1
        if (issuer) {
            if (t + 1 < 256) {
                const unsigned st = lds_base + ((unsigned)(t + 1) & 1u) * STAGE_B;
                tdm_load_tile(w_hi + (size_t)(t + 1) * 16 * KDIM, st);
                tdm_load_tile(w_lo + (size_t)(t + 1) * 16 * KDIM, st + TILE_B);
                __builtin_amdgcn_s_wait_tensorcnt(2);  // tile t done; t+1 in flight
            } else {
                __builtin_amdgcn_s_wait_tensorcnt(0);
            }
        }
        __syncthreads();   // publish tile t to all waves

        const char* hiB = &ldsb[t & 1][0];
        const char* loB = &ldsb[t & 1][TILE_B];
        const int col = t * 16 + l15;
        const float wn = wnorm[col];
        v8f acc = {0.f, 0.f, 0.f, 0.f, 0.f, 0.f, 0.f, 0.f};
#pragma unroll
        for (int ks = 0; ks < 8; ++ks) {
            const uint4* ph = (const uint4*)(hiB + cOff + ks * 64);
            const uint4* pl = (const uint4*)(loB + cOff + ks * 64);
            V16 tb;
            tb.u[0] = ph[0]; tb.u[1] = ph[2];
            v16bf bh = tb.v;
            tb.u[0] = pl[0]; tb.u[1] = pl[2];
            v16bf bl = tb.v;
            acc = __builtin_amdgcn_wmma_f32_16x16x32_bf16(false, a_hi[ks], false, bh,
                                                          (short)0, acc, false, false);
            acc = __builtin_amdgcn_wmma_f32_16x16x32_bf16(false, a_hi[ks], false, bl,
                                                          (short)0, acc, false, false);
            acc = __builtin_amdgcn_wmma_f32_16x16x32_bf16(false, a_lo[ks], false, bh,
                                                          (short)0, acc, false, false);
        }
#pragma unroll
        for (int i = 0; i < 8; ++i) {
            float s = wn - 2.0f * acc[i];
            if (s < bestVal[i]) { bestVal[i] = s; bestIdx[i] = col; }
        }
    }

    // Reduce across the 16 lanes holding the same rows (xor stays in each half).
#pragma unroll
    for (int i = 0; i < 8; ++i) {
        float v = bestVal[i];
        int   id = bestIdx[i];
#pragma unroll
        for (int m = 1; m < 16; m <<= 1) {
            float ov = __shfl_xor(v, m, 32);
            int   oi = __shfl_xor(id, m, 32);
            if (ov < v || (ov == v && oi < id)) { v = ov; id = oi; }
        }
        if (l15 == 0) {
            int row = r0 + h * 8 + i;      // C/D layout: M = i + 8*h
            idx_out[row] = id;
            enc_out[row] = (float)id;
        }
    }
}

// ---------------------------------------------------------------------------
// 4) Gather codebook rows into quant_feat / quant_feat_st, reduce MSE.
// ---------------------------------------------------------------------------
__global__ void gather_kernel(const float* __restrict__ grid,
                              const float* __restrict__ weight,
                              const int* __restrict__ idx,
                              float* __restrict__ out,
                              float* __restrict__ sumsq) {
    __shared__ float red[256];
    const size_t t = (size_t)blockIdx.x * 256 + threadIdx.x;   // 0..QELEMS-1
    const int s  = (int)(t & 4095);
    const int bc = (int)(t >> 12);
    const int c  = bc & 255;
    const int b  = bc >> 8;
    const int n  = (b << 12) | s;
    const float q = weight[(size_t)idx[n] * KDIM + c];
    out[t]          = q;   // quant_feat
    out[t + QELEMS] = q;   // quant_feat_st (numerically identical forward)
    const float d = grid[t] - q;
    red[threadIdx.x] = d * d;
    __syncthreads();
    for (int off = 128; off > 0; off >>= 1) {
        if (threadIdx.x < off) red[threadIdx.x] += red[threadIdx.x + off];
        __syncthreads();
    }
    if (threadIdx.x == 0) atomicAdd(sumsq, red[0]);
}

__global__ void finalize_kernel(const float* __restrict__ sumsq,
                                float* __restrict__ out) {
    out[0] = sumsq[0] / (float)QELEMS;
}

// ---------------------------------------------------------------------------
extern "C" void kernel_launch(void* const* d_in, const int* in_sizes, int n_in,
                              void* d_out, int out_size, void* d_ws, size_t ws_size,
                              hipStream_t stream) {
    (void)in_sizes; (void)n_in; (void)out_size; (void)ws_size;
    const float* grid   = (const float*)d_in[0];   // (8,256,16,16,16) f32
    const float* weight = (const float*)d_in[1];   // (4096,256) f32
    float* out = (float*)d_out;

    char* ws = (char*)d_ws;
    unsigned short* f_hi = (unsigned short*)(ws);
    unsigned short* f_lo = (unsigned short*)(ws + 16777216);
    unsigned short* w_hi = (unsigned short*)(ws + 2 * 16777216);
    unsigned short* w_lo = (unsigned short*)(ws + 2 * 16777216 + 2097152);
    float* wnorm = (float*)(ws + 2 * 16777216 + 2 * 2097152);
    int*   idx   = (int*)  (ws + 2 * 16777216 + 2 * 2097152 + 16384);
    float* sumsq = (float*)(ws + 2 * 16777216 + 2 * 2097152 + 16384 + 131072);

    prep_weight  <<<NCODES,        256, 0, stream>>>(weight, w_hi, w_lo, wnorm, sumsq);
    prep_feat    <<<8192,          256, 0, stream>>>(grid, f_hi, f_lo);
    argmin_gemm  <<<NROWS/128,     256, 0, stream>>>(f_hi, f_lo, w_hi, w_lo, wnorm,
                                                     idx, out + ENC_OFF);
    gather_kernel<<<QELEMS/256,    256, 0, stream>>>(grid, weight, idx, out, sumsq);
    finalize_kernel<<<1, 1, 0, stream>>>(sumsq, out + DIFF_OFF);
}
